// Seq2Seq_53437983096942
// MI455X (gfx1250) — matrix operand order repaired and checked by reading, not compiled, not run on hardware
//
#include <hip/hip_runtime.h>
#include <math.h>

// ---------------------------------------------------------------------------
// Problem constants (from reference)
// ---------------------------------------------------------------------------
#define BATCH 64
#define TIN   128
#define NIN   256
#define MH    1024
#define OUTD  512
#define LM    2048
#define TDEC  64
#define G3    (3*MH)          // 3072 gate width

// ---------------------------------------------------------------------------
// types
// ---------------------------------------------------------------------------
typedef __attribute__((ext_vector_type(16))) __bf16        v16bf;
typedef __attribute__((ext_vector_type(8)))  float         v8f;
typedef __attribute__((ext_vector_type(4)))  unsigned int  v4u;
typedef __attribute__((ext_vector_type(8)))  int           v8i;
typedef __attribute__((ext_vector_type(4)))  int           v4i;

__device__ __forceinline__ float bf2f(unsigned short h) {
    union { unsigned int u; float f; } c; c.u = ((unsigned int)h) << 16; return c.f;
}
__device__ __forceinline__ unsigned short f2bf(float x) {
    union { float f; unsigned int u; } c; c.f = x;
    unsigned int r = c.u + 0x7FFFu + ((c.u >> 16) & 1u);
    return (unsigned short)(r >> 16);
}

// ---------------------------------------------------------------------------
// Generic cast / zero kernels
// ---------------------------------------------------------------------------
__global__ void cast_f32_bf16_kernel(const float* __restrict__ src,
                                     unsigned short* __restrict__ dst, int n) {
    int i = blockIdx.x * blockDim.x + threadIdx.x;
    if (i < n) dst[i] = f2bf(src[i]);
}
__global__ void zero_bf16_kernel(unsigned short* __restrict__ p, int n) {
    int i = blockIdx.x * blockDim.x + threadIdx.x;
    if (i < n) p[i] = 0;
}

// ---------------------------------------------------------------------------
// Tensor Data Mover: DMA a 16 x K bf16 strip (row stride lda elems) into LDS.
// D# packed per CDNA5 ISA 8.3/8.4: 2D tile, data_size=2B, type=2, count=1.
// Issued by one wave; completion via s_wait_tensorcnt.
// This toolchain's builtin is the 6-arg form:
//   (uint32x4 g0, int32x8 g1, int32x4 g2, int32x4 g3, int32x8 g4, i32 cpol)
// Groups 2/3/4 are zero (2-D tensor).
// ---------------------------------------------------------------------------
__device__ __forceinline__ void tdm_load_strip(const unsigned short* gsrc,
                                               unsigned ldsOff, int K, int lda) {
    unsigned long long ga = (unsigned long long)(__UINTPTR_TYPE__)gsrc;
    v4u g0;
    g0[0] = 1u;                                                // count=1 (user D#)
    g0[1] = ldsOff;                                            // lds_addr (bytes)
    g0[2] = (unsigned)(ga & 0xFFFFFFFFull);                    // global_addr[31:0]
    g0[3] = (unsigned)((ga >> 32) & 0x01FFFFFFull) | (2u << 30); // ga[56:32] | type=2

    const unsigned td0 = (unsigned)lda;     // tensor_dim0 (elements)
    const unsigned td1 = 16u;               // tensor_dim1
    const unsigned tl0 = (unsigned)K;       // tile_dim0
    const unsigned tl1 = 16u;               // tile_dim1 (tile_dim2=0)
    const unsigned long long s0 = (unsigned long long)lda;  // tensor_dim0_stride

    v8i g1;
    g1[0] = (int)(1u << 16);                                   // wg_mask=0, data_size=2B
    g1[1] = (int)((td0 & 0xFFFFu) << 16);                      // atomic_bar=0 | td0 lo
    g1[2] = (int)(((td0 >> 16) & 0xFFFFu) | ((td1 & 0xFFFFu) << 16));
    g1[3] = (int)(((td1 >> 16) & 0xFFFFu) | (tl0 << 16));      // tile_dim0
    g1[4] = (int)(tl1);                                        // tile_dim1 | tile_dim2=0
    g1[5] = (int)(unsigned)(s0 & 0xFFFFFFFFull);               // stride0 lo32
    g1[6] = (int)((unsigned)((s0 >> 32) & 0xFFFFu));           // stride0 hi16 | stride1 lo=0
    g1[7] = 0;
    v4i g2 = {0, 0, 0, 0}, g3 = {0, 0, 0, 0};
    v8i g4 = {0, 0, 0, 0, 0, 0, 0, 0};
    __builtin_amdgcn_tensor_load_to_lds(g0, g1, g2, g3, g4, 0);
}

// ---------------------------------------------------------------------------
// WMMA bf16 GEMM:  C[rows x cols] = A[rows x K] @ W[cols x K]^T + bias
// Block = 8 waves, one 16(M) x 512(N) super-strip; A strip staged in LDS by TDM.
// Requires cols % 512 == 0 (true for 3072 / 2048 / 512) so every wave is live
// and EXEC is all-1s for WMMA.
// Fragment layouts per CDNA5 ISA 7.12.2 (wave32):
//   A 16x32: lane m=L&15; two 16B chunks at k = kk+8*(L>>4) and +16
//   B 32x16: lane n=L&15; VGPR v holds K = kk + 2*(v&3) + 16*(v>>2) + 8*(L>>4)
//   C 16x16: elem i of lane L -> row = i + 8*(L>>4), col = L&15
// ---------------------------------------------------------------------------
union AFrag { v16bf v; unsigned int u[8]; };

__global__ void gemm_bf16_wmma_kernel(const unsigned short* __restrict__ A, int lda,
                                      const unsigned short* __restrict__ W,
                                      const float* __restrict__ bias,
                                      float* __restrict__ C, int ldc,
                                      int cols, int K) {
    extern __shared__ unsigned short smem[];        // 16 x K bf16 A-strip
    const int lane = threadIdx.x & 31;
    const int wv   = threadIdx.x >> 5;              // 0..7
    const int mt   = blockIdx.x;                    // 16-row strip index
    const int nt   = blockIdx.y * 8 + wv;           // 64-col tile index (always valid)
    const int hi   = lane >> 4;
    const int lo   = lane & 15;

    if (threadIdx.x < 32)                           // wave 0 issues the DMA
        tdm_load_strip(A + (size_t)mt * 16 * lda, 0u, K, lda);
    __builtin_amdgcn_s_wait_tensorcnt(0);
    __syncthreads();

    const unsigned short* W0 = W + (size_t)(nt * 64 +  0 + lo) * K;
    const unsigned short* W1 = W + (size_t)(nt * 64 + 16 + lo) * K;
    const unsigned short* W2 = W + (size_t)(nt * 64 + 32 + lo) * K;
    const unsigned short* W3 = W + (size_t)(nt * 64 + 48 + lo) * K;
    const unsigned short* arow0 = smem + (size_t)lo * K + 8 * hi;

    v8f acc0 = {}, acc1 = {}, acc2 = {}, acc3 = {};

    for (int kk = 0; kk < K; kk += 32) {
        AFrag a;
        const unsigned short* ar = arow0 + kk;
#pragma unroll
        for (int v = 0; v < 4; ++v) {               // -> ds_load_b128 x2
            a.u[v]     = *(const unsigned int*)(ar + 2 * v);
            a.u[4 + v] = *(const unsigned int*)(ar + 16 + 2 * v);
        }
        if (kk + 32 < K) __builtin_prefetch(W0 + kk + 32, 0, 1);   // global_prefetch_b8

        AFrag b0, b1, b2, b3;
        const int kb = kk + 8 * hi;
#pragma unroll
        for (int v = 0; v < 8; ++v) {
            int k = kb + 2 * (v & 3) + 16 * (v >> 2);
            b0.u[v] = *(const unsigned int*)(W0 + k);
            b1.u[v] = *(const unsigned int*)(W1 + k);
            b2.u[v] = *(const unsigned int*)(W2 + k);
            b3.u[v] = *(const unsigned int*)(W3 + k);
        }
        acc0 = __builtin_amdgcn_wmma_f32_16x16x32_bf16(false, a.v, false, b0.v, (short)0, acc0, false, false);
        acc1 = __builtin_amdgcn_wmma_f32_16x16x32_bf16(false, a.v, false, b1.v, (short)0, acc1, false, false);
        acc2 = __builtin_amdgcn_wmma_f32_16x16x32_bf16(false, a.v, false, b2.v, (short)0, acc2, false, false);
        acc3 = __builtin_amdgcn_wmma_f32_16x16x32_bf16(false, a.v, false, b3.v, (short)0, acc3, false, false);
    }

#pragma unroll
    for (int i = 0; i < 8; ++i) {
        const int row = mt * 16 + i + 8 * hi;
        float* crow = C + (size_t)row * ldc + nt * 64 + lo;
        const float* brow = bias + nt * 64 + lo;
        crow[ 0] = acc0[i] + brow[ 0];
        crow[16] = acc1[i] + brow[16];
        crow[32] = acc2[i] + brow[32];
        crow[48] = acc3[i] + brow[48];
    }
}

// ---------------------------------------------------------------------------
// Fused GRU pointwise combine:
//   r = sig(gi_r+gh_r); z = sig(gi_z+gh_z); n = tanh(gi_n + r*gh_n)
//   h' = (1-z)*n + z*h      (h read/written as bf16)
// ---------------------------------------------------------------------------
__global__ void gru_combine_kernel(const float* __restrict__ GI, long long gi_stride,
                                   const float* __restrict__ GH,
                                   const unsigned short* __restrict__ hprev,
                                   unsigned short* __restrict__ hnew) {
    int idx = blockIdx.x * blockDim.x + threadIdx.x;      // BATCH*MH threads
    int b = idx >> 10, m = idx & (MH - 1);
    const float* gi = GI + (long long)b * gi_stride;
    const float* gh = GH + (long long)b * G3;
    float ir = gi[m], iz = gi[MH + m], in = gi[2 * MH + m];
    float hr = gh[m], hz = gh[MH + m], hn = gh[2 * MH + m];
    float r = 1.0f / (1.0f + expf(-(ir + hr)));
    float z = 1.0f / (1.0f + expf(-(iz + hz)));
    float n = tanhf(in + r * hn);
    float h = bf2f(hprev[(size_t)b * MH + m]);
    hnew[(size_t)b * MH + m] = f2bf((1.0f - z) * n + z * h);
}

// ---------------------------------------------------------------------------
// LayerNorm + exact GELU over rows of LM=2048; out bf16 (feeds FC2 WMMA)
// ---------------------------------------------------------------------------
__global__ void ln_gelu_kernel(const float* __restrict__ Z,
                               const float* __restrict__ gamma,
                               const float* __restrict__ beta,
                               unsigned short* __restrict__ out) {
    __shared__ float s1[256], s2[256];
    const int b = blockIdx.x, tid = threadIdx.x;
    const float* z = Z + (size_t)b * LM;
    float a = 0.f, q = 0.f;
    for (int i = tid; i < LM; i += 256) { float v = z[i]; a += v; q += v * v; }
    s1[tid] = a; s2[tid] = q; __syncthreads();
    for (int s = 128; s > 0; s >>= 1) {
        if (tid < s) { s1[tid] += s1[tid + s]; s2[tid] += s2[tid + s]; }
        __syncthreads();
    }
    const float mean = s1[0] * (1.0f / LM);
    const float var  = s2[0] * (1.0f / LM) - mean * mean;
    const float inv  = rsqrtf(var + 1e-5f);
    for (int i = tid; i < LM; i += 256) {
        float y = (z[i] - mean) * inv * gamma[i] + beta[i];
        float g = 0.5f * y * (1.0f + erff(y * 0.70710678118654752f));
        out[(size_t)b * LM + i] = f2bf(g);
    }
}

// ---------------------------------------------------------------------------
// Softmax over O=512 logits (strided rows inside d_out) -> bf16 feedback y
// ---------------------------------------------------------------------------
__global__ void softmax_kernel(const float* __restrict__ logits, long long row_stride,
                               unsigned short* __restrict__ y) {
    __shared__ float red[256];
    const int b = blockIdx.x, tid = threadIdx.x;
    const float* L = logits + (long long)b * row_stride;
    float mx = -3.4e38f;
    for (int i = tid; i < OUTD; i += 256) mx = fmaxf(mx, L[i]);
    red[tid] = mx; __syncthreads();
    for (int s = 128; s > 0; s >>= 1) { if (tid < s) red[tid] = fmaxf(red[tid], red[tid + s]); __syncthreads(); }
    mx = red[0]; __syncthreads();
    float sum = 0.f;
    for (int i = tid; i < OUTD; i += 256) sum += expf(L[i] - mx);
    red[tid] = sum; __syncthreads();
    for (int s = 128; s > 0; s >>= 1) { if (tid < s) red[tid] += red[tid + s]; __syncthreads(); }
    const float rinv = 1.0f / red[0];
    for (int i = tid; i < OUTD; i += 256) y[(size_t)b * OUTD + i] = f2bf(expf(L[i] - mx) * rinv);
}

// ---------------------------------------------------------------------------
// Host side
// ---------------------------------------------------------------------------
static inline void launch_cast(const float* s, unsigned short* d, int n, hipStream_t st) {
    cast_f32_bf16_kernel<<<(n + 255) / 256, 256, 0, st>>>(s, d, n);
}
static inline void launch_gemm(const unsigned short* A, int lda, const unsigned short* W,
                               const float* bias, float* C, int ldc,
                               int rows, int cols, int K, hipStream_t st) {
    dim3 grid(rows >> 4, cols >> 9);            // cols % 512 == 0 for all call sites
    size_t shmem = (size_t)32 * (size_t)K;      // 16 rows x K bf16
    gemm_bf16_wmma_kernel<<<grid, 256, shmem, st>>>(A, lda, W, bias, C, ldc, cols, K);
}

extern "C" void kernel_launch(void* const* d_in, const int* in_sizes, int n_in,
                              void* d_out, int out_size, void* d_ws, size_t ws_size,
                              hipStream_t stream) {
    (void)in_sizes; (void)n_in; (void)out_size; (void)ws_size;

    const float* X        = (const float*)d_in[0];
    const float* Y0       = (const float*)d_in[1];
    const float* eWih0    = (const float*)d_in[2];
    const float* eWhh0    = (const float*)d_in[3];
    const float* ebih0    = (const float*)d_in[4];
    const float* ebhh0    = (const float*)d_in[5];
    const float* eWih1    = (const float*)d_in[6];
    const float* eWhh1    = (const float*)d_in[7];
    const float* ebih1    = (const float*)d_in[8];
    const float* ebhh1    = (const float*)d_in[9];
    const float* dWih0    = (const float*)d_in[10];
    const float* dWhh0    = (const float*)d_in[11];
    const float* dbih0    = (const float*)d_in[12];
    const float* dbhh0    = (const float*)d_in[13];
    const float* dWih1    = (const float*)d_in[14];
    const float* dWhh1    = (const float*)d_in[15];
    const float* dbih1    = (const float*)d_in[16];
    const float* dbhh1    = (const float*)d_in[17];
    const float* fc1w     = (const float*)d_in[18];
    const float* fc1b     = (const float*)d_in[19];
    const float* lng      = (const float*)d_in[20];
    const float* lnb      = (const float*)d_in[21];
    const float* fc2w     = (const float*)d_in[22];
    const float* fc2b     = (const float*)d_in[23];
    float* out = (float*)d_out;

    // -------- workspace carve-up (256B aligned) --------
    char* base = (char*)d_ws;
    size_t off = 0;
    auto carve = [&](size_t bytes) -> char* {
        char* p = base + off;
        off += (bytes + 255) & ~(size_t)255;
        return p;
    };
    unsigned short* bWih0  = (unsigned short*)carve((size_t)G3 * NIN  * 2);
    unsigned short* bWhh0  = (unsigned short*)carve((size_t)G3 * MH   * 2);
    unsigned short* bWih1  = (unsigned short*)carve((size_t)G3 * MH   * 2);
    unsigned short* bWhh1  = (unsigned short*)carve((size_t)G3 * MH   * 2);
    unsigned short* bdWih0 = (unsigned short*)carve((size_t)G3 * OUTD * 2);
    unsigned short* bdWhh0 = (unsigned short*)carve((size_t)G3 * MH   * 2);
    unsigned short* bdWih1 = (unsigned short*)carve((size_t)G3 * MH   * 2);
    unsigned short* bdWhh1 = (unsigned short*)carve((size_t)G3 * MH   * 2);
    unsigned short* bfc1   = (unsigned short*)carve((size_t)LM * MH   * 2);
    unsigned short* bfc2   = (unsigned short*)carve((size_t)OUTD * LM * 2);
    unsigned short* Xbf    = (unsigned short*)carve((size_t)BATCH * TIN * NIN * 2);
    unsigned short* H0seq  = (unsigned short*)carve((size_t)TIN * BATCH * MH * 2);
    unsigned short* h0zero = (unsigned short*)carve((size_t)BATCH * MH * 2);
    unsigned short* h1st   = (unsigned short*)carve((size_t)BATCH * MH * 2);
    unsigned short* h0d    = (unsigned short*)carve((size_t)BATCH * MH * 2);
    unsigned short* ybf    = (unsigned short*)carve((size_t)BATCH * OUTD * 2);
    unsigned short* zbf    = (unsigned short*)carve((size_t)BATCH * LM * 2);
    float* GI0  = (float*)carve((size_t)BATCH * TIN * G3 * 4);   // 96 MB
    float* GIb  = (float*)carve((size_t)BATCH * G3 * 4);
    float* GHb  = (float*)carve((size_t)BATCH * G3 * 4);
    float* Zbuf = (float*)carve((size_t)BATCH * LM * 4);

    // -------- cast weights + inputs to bf16 --------
    launch_cast(eWih0, bWih0,  G3 * NIN,  stream);
    launch_cast(eWhh0, bWhh0,  G3 * MH,   stream);
    launch_cast(eWih1, bWih1,  G3 * MH,   stream);
    launch_cast(eWhh1, bWhh1,  G3 * MH,   stream);
    launch_cast(dWih0, bdWih0, G3 * OUTD, stream);
    launch_cast(dWhh0, bdWhh0, G3 * MH,   stream);
    launch_cast(dWih1, bdWih1, G3 * MH,   stream);
    launch_cast(dWhh1, bdWhh1, G3 * MH,   stream);
    launch_cast(fc1w,  bfc1,   LM * MH,   stream);
    launch_cast(fc2w,  bfc2,   OUTD * LM, stream);
    launch_cast(X,     Xbf,    BATCH * TIN * NIN, stream);
    launch_cast(Y0,    ybf,    BATCH * OUTD, stream);
    zero_bf16_kernel<<<(BATCH * MH + 255) / 256, 256, 0, stream>>>(h0zero, BATCH * MH);
    zero_bf16_kernel<<<(BATCH * MH + 255) / 256, 256, 0, stream>>>(h1st,   BATCH * MH);

    const int nComb = BATCH * MH;   // 65536 threads for GRU combine

    // -------- encoder layer 0: hoisted input GEMM (8192 x 3072 x 256) --------
    launch_gemm(Xbf, NIN, bWih0, ebih0, GI0, G3, BATCH * TIN, G3, NIN, stream);

    // sequential h-side recurrence; h_t written into H0seq[t] (bf16)
    for (int t = 0; t < TIN; ++t) {
        const unsigned short* hprev = (t == 0) ? h0zero : (H0seq + (size_t)(t - 1) * BATCH * MH);
        launch_gemm(hprev, MH, bWhh0, ebhh0, GHb, G3, BATCH, G3, MH, stream);
        gru_combine_kernel<<<nComb / 256, 256, 0, stream>>>(
            GI0 + (size_t)t * G3, (long long)TIN * G3, GHb, hprev,
            H0seq + (size_t)t * BATCH * MH);
    }

    // -------- encoder layer 1 (per-step input + hidden GEMMs) --------
    for (int t = 0; t < TIN; ++t) {
        launch_gemm(H0seq + (size_t)t * BATCH * MH, MH, bWih1, ebih1, GIb, G3, BATCH, G3, MH, stream);
        launch_gemm(h1st, MH, bWhh1, ebhh1, GHb, G3, BATCH, G3, MH, stream);
        gru_combine_kernel<<<nComb / 256, 256, 0, stream>>>(GIb, (long long)G3, GHb, h1st, h1st);
    }

    // decoder initial states: h0d = encoder L0 final hidden; h1st already L1 final
    (void)hipMemcpyAsync(h0d, H0seq + (size_t)(TIN - 1) * BATCH * MH,
                         (size_t)BATCH * MH * 2, hipMemcpyDeviceToDevice, stream);

    // -------- decoder: 64 autoregressive steps --------
    const long long ldo = (long long)TDEC * OUTD;   // row stride of d_out (B, TDEC, O)
    for (int s = 0; s < TDEC; ++s) {
        // GRU cell 0 (input = softmax feedback y, O=512)
        launch_gemm(ybf, OUTD, bdWih0, dbih0, GIb, G3, BATCH, G3, OUTD, stream);
        launch_gemm(h0d, MH,   bdWhh0, dbhh0, GHb, G3, BATCH, G3, MH,   stream);
        gru_combine_kernel<<<nComb / 256, 256, 0, stream>>>(GIb, (long long)G3, GHb, h0d, h0d);
        // GRU cell 1 (input = h0d)
        launch_gemm(h0d,  MH, bdWih1, dbih1, GIb, G3, BATCH, G3, MH, stream);
        launch_gemm(h1st, MH, bdWhh1, dbhh1, GHb, G3, BATCH, G3, MH, stream);
        gru_combine_kernel<<<nComb / 256, 256, 0, stream>>>(GIb, (long long)G3, GHb, h1st, h1st);
        // FC1 -> LN -> GELU
        launch_gemm(h1st, MH, bfc1, fc1b, Zbuf, LM, BATCH, LM, MH, stream);
        ln_gelu_kernel<<<BATCH, 256, 0, stream>>>(Zbuf, lng, lnb, zbf);
        // FC2 writes logits straight into d_out[:, s, :]
        launch_gemm(zbf, LM, bfc2, fc2b, out + (size_t)s * OUTD, (int)ldo, BATCH, OUTD, LM, stream);
        // softmax feedback
        softmax_kernel<<<BATCH, 256, 0, stream>>>(out + (size_t)s * OUTD, ldo, ybf);
    }
}
